// WirelessSplitNet_1468878815382
// MI455X (gfx1250) — matrix operand order repaired and checked by reading, not compiled, not run on hardware
//
#include <hip/hip_runtime.h>
#include <hip/hip_bf16.h>

// ---------------------------------------------------------------------------
// WirelessSplitNet forward, fused for MI455X (gfx1250, wave32, WMMA bf16,
// async global->LDS DMA staging).
//
// Pipeline (all on `stream`):
//   k_zero   : zero accumulators (R, S1/S2, cA/cB, scalars) in d_ws
//   k_conv   : pre-convert x and W1 to bf16, K padded 98->128 (zero pad),
//              so GEMM tiles can be staged with GLOBAL_LOAD_ASYNC_TO_LDS_B128
//   k_prep   : per-(d,j) channel gains w[d,j], fused bf16 weight
//              Wf[(d*32+j)*2048+o] = w[d,j]*Wc[d,j,o], constants cA[j], cB[j]
//   k_gemm1  : sub = relu(x_d @ W1_d^T + b1) via v_wmma_f32_16x16x32_bf16,
//              async-DMA staged tiles; bf16 sub + per-device sum/sumsq stats
//   k_stats  : gm, inv_s = rsqrt(gv+1e-6) scalars
//   k_gemm2  : R[b,j] = sum_{d,o} sub * Wf  (split-K over d, f32 atomics),
//              async-DMA staged A and B tiles
//   k_final  : received = inv_s*R + const; relu; MLP 32->98->49->10; log_softmax
// ---------------------------------------------------------------------------

typedef __attribute__((ext_vector_type(8)))  float          v8f;
typedef __attribute__((ext_vector_type(16))) __bf16         v16bf;
typedef __attribute__((ext_vector_type(16))) unsigned short v16u;

#define BATCH   1000
#define MPAD    1024
#define NDEV    8
#define PRE_K   98
#define KPAD    128
#define NXT     2048
#define NSYM    32
#define KTOT    16384   // NDEV*NXT

#define XE      (MPAD * NDEV * KPAD)   // padded bf16 x elements
#define WE      (NDEV * NXT * KPAD)    // padded bf16 W1 elements

__device__ __forceinline__ unsigned short f2bf(float x) {
    unsigned int u = __float_as_uint(x);
    u += 0x7FFFu + ((u >> 16) & 1u);   // round-to-nearest-even
    return (unsigned short)(u >> 16);
}

// Async DMA: global (saddr + 32-bit voffset) -> LDS (16 bytes), ASYNCcnt.
__device__ __forceinline__ void async_b128(unsigned lds_off, const void* sbase,
                                           unsigned goff) {
    asm volatile("global_load_async_to_lds_b128 %0, %1, %2"
                 :: "v"(lds_off), "v"(goff), "s"(sbase) : "memory");
}
__device__ __forceinline__ void async_wait0() {
    asm volatile("s_wait_asynccnt 0" ::: "memory");
}

// ------------------------------ zero scratch -------------------------------
__global__ void k_zero(float* __restrict__ p, int n) {
    int i = blockIdx.x * blockDim.x + threadIdx.x;
    if (i < n) p[i] = 0.0f;
}

// ------------------------------ bf16 pre-convert (K padded to 128) ---------
__global__ void k_conv(const float* __restrict__ x, const float* __restrict__ W1,
                       unsigned short* __restrict__ xbf,
                       unsigned short* __restrict__ w1bf) {
    int i = blockIdx.x * blockDim.x + threadIdx.x;
    if (i < XE) {
        int k = i & 127, bd = i >> 7;
        int d = bd & 7, b = bd >> 3;             // xbf[b][d][128]
        float v = (k < PRE_K && b < BATCH) ? x[b * 784 + d * PRE_K + k] : 0.0f;
        xbf[i] = f2bf(v);
    } else if (i < XE + WE) {
        int t = i - XE;
        int k = t & 127, nd = t >> 7;            // w1bf[d*2048+n][128]
        float v = (k < PRE_K) ? W1[(size_t)nd * PRE_K + k] : 0.0f;
        w1bf[t] = f2bf(v);
    }
}

// ------------------------------ prep: fused channel weights ----------------
// grid: 256 blocks -> (d,j). block: 64 threads. Wf layout: [(d*32+j)][o].
__global__ __launch_bounds__(64) void k_prep(
    const float* __restrict__ Wc, const float* __restrict__ bc,
    const float* __restrict__ cut, const float* __restrict__ b_mat,
    const float* __restrict__ h_mat, const float* __restrict__ a_mat,
    const float* __restrict__ noise, const float* __restrict__ indicator,
    unsigned short* __restrict__ Wf, float* __restrict__ cAB)
{
    __shared__ float wsh;
    __shared__ float red[64];
    const int d = blockIdx.x >> 5;
    const int j = blockIdx.x & 31;
    const int tid = threadIdx.x;

    if (tid == 0) {
        float hv = 0.0f;
        for (int a = 0; a < 5; ++a) {
            float hs = 0.0f;
            for (int k = 0; k < 64; ++k)
                hs += indicator[j * 64 + k] * h_mat[(d * 64 + k) * 5 + a];
            hv += hs * a_mat[j * 5 + a];
        }
        wsh = b_mat[d * 32 + j] * hv;
    }
    __syncthreads();
    const float w = wsh;

    float ps = 0.0f;
    for (int o = tid; o < NXT; o += 64) {
        float wc = Wc[((size_t)d * 32 + j) * NXT + o];
        ps += wc;
        Wf[((size_t)(d * 32 + j)) * NXT + o] = f2bf(w * wc);
    }
    red[tid] = ps;
    __syncthreads();
    for (int s = 32; s > 0; s >>= 1) {
        if (tid < s) red[tid] += red[tid + s];
        __syncthreads();
    }
    if (tid == 0) {
        float wcsum = red[0];
        float ca = w * bc[d * 32 + j];
        if (d == 0) {
            float ns = 0.0f;
            for (int a = 0; a < 5; ++a) ns += a_mat[j * 5 + a] * noise[j * 5 + a];
            ca += ns + cut[j];
        }
        atomicAdd(&cAB[j], ca);              // cA[j]
        atomicAdd(&cAB[32 + j], w * wcsum);  // cB[j]
    }
}

// ------------------------------ GEMM1: sub = relu(x W1^T + b1) -------------
// grid: (8 Mtiles, 32 Ntiles, 8 devices). block: 256 (8 waves, 4x2 wave grid).
// Tiles staged by async DMA from pre-converted bf16 buffers.
__global__ __launch_bounds__(256) void k_gemm1(
    const unsigned short* __restrict__ xbf, const unsigned short* __restrict__ w1bf,
    const float* __restrict__ b1,
    unsigned short* __restrict__ sub, float* __restrict__ S12)
{
    __shared__ unsigned short Xs[128 * KPAD];  // 32 KB
    __shared__ unsigned short Ws[64 * KPAD];   // 16 KB
    __shared__ float redA[256], redB[256];

    const int d  = blockIdx.z;
    const int m0 = blockIdx.x * 128;
    const int n0 = blockIdx.y * 64;
    const int tid = threadIdx.x;

    // X tile: 128 rows x 256B = 2048 x 16B transfers, 8 per thread
#pragma unroll
    for (int t = 0; t < 8; ++t) {
        int idx = t * 256 + tid;
        int r = idx >> 4, g = idx & 15;
        async_b128((unsigned)(uintptr_t)&Xs[r * KPAD + g * 8], xbf,
                   (unsigned)((((m0 + r) * NDEV + d) * KPAD + g * 8) * 2));
    }
    // W tile: 64 rows x 256B = 1024 x 16B transfers, 4 per thread
#pragma unroll
    for (int t = 0; t < 4; ++t) {
        int idx = t * 256 + tid;
        int r = idx >> 4, g = idx & 15;
        async_b128((unsigned)(uintptr_t)&Ws[r * KPAD + g * 8], w1bf,
                   (unsigned)(((d * NXT + n0 + r) * KPAD + g * 8) * 2));
    }
    async_wait0();
    __syncthreads();

    const int wave = tid >> 5, lane = tid & 31;
    const int half = lane >> 4, l16 = lane & 15;
    const int wm = (wave & 3) * 32;
    const int wn = (wave >> 2) * 32;

    v8f acc[2][2] = {};
    for (int ks = 0; ks < KPAD; ks += 32) {
        v16u a[2], bm[2];
#pragma unroll
        for (int mi = 0; mi < 2; ++mi) {
            int row = wm + mi * 16 + l16;
#pragma unroll
            for (int v = 0; v < 8; ++v) {
                int kk = ks + ((v < 4) ? (8 * half + 2 * v)
                                       : (16 + 8 * half + 2 * (v - 4)));
                unsigned int t = *(const unsigned int*)&Xs[row * KPAD + kk];
                a[mi][2 * v]     = (unsigned short)t;
                a[mi][2 * v + 1] = (unsigned short)(t >> 16);
            }
        }
#pragma unroll
        for (int ni = 0; ni < 2; ++ni) {
            int col = wn + ni * 16 + l16;
#pragma unroll
            for (int v = 0; v < 8; ++v) {
                int kk = ks + ((v < 4) ? (8 * half + 2 * v)
                                       : (16 + 8 * half + 2 * (v - 4)));
                unsigned int t = *(const unsigned int*)&Ws[col * KPAD + kk];
                bm[ni][2 * v]     = (unsigned short)t;
                bm[ni][2 * v + 1] = (unsigned short)(t >> 16);
            }
        }
#pragma unroll
        for (int mi = 0; mi < 2; ++mi)
#pragma unroll
            for (int ni = 0; ni < 2; ++ni)
                acc[mi][ni] = __builtin_amdgcn_wmma_f32_16x16x32_bf16(
                    false, __builtin_bit_cast(v16bf, a[mi]),
                    false, __builtin_bit_cast(v16bf, bm[ni]),
                    (short)0, acc[mi][ni], false, false);
    }

    // epilogue: bias + relu + bf16 store + per-device stats
    float s1 = 0.0f, s2 = 0.0f;
#pragma unroll
    for (int mi = 0; mi < 2; ++mi) {
#pragma unroll
        for (int ni = 0; ni < 2; ++ni) {
            int n = n0 + wn + ni * 16 + l16;
            float bias = b1[d * NXT + n];
#pragma unroll
            for (int r = 0; r < 8; ++r) {
                int m = m0 + wm + mi * 16 + 8 * half + r;
                float v = acc[mi][ni][r] + bias;
                v = v > 0.0f ? v : 0.0f;
                if (m < BATCH) { s1 += v; s2 += v * v; }
                sub[(size_t)m * KTOT + d * NXT + n] = f2bf(v);
            }
        }
    }
    redA[tid] = s1; redB[tid] = s2;
    __syncthreads();
    for (int s = 128; s > 0; s >>= 1) {
        if (tid < s) { redA[tid] += redA[tid + s]; redB[tid] += redB[tid + s]; }
        __syncthreads();
    }
    if (tid == 0) {
        atomicAdd(&S12[d], redA[0]);
        atomicAdd(&S12[8 + d], redB[0]);
    }
}

// ------------------------------ stats finalize -----------------------------
__global__ void k_stats(const float* __restrict__ S12, float* __restrict__ sc) {
    if (threadIdx.x == 0) {
        const float inv = 1.0f / ((float)BATCH * (float)NXT);
        float gm = 0.0f, gv = 0.0f;
        for (int d = 0; d < NDEV; ++d) {
            float m = S12[d] * inv;
            float v = S12[8 + d] * inv - m * m;
            gm += m; gv += v;
        }
        gm *= 0.125f; gv *= 0.125f;
        sc[0] = gm;
        sc[1] = rsqrtf(gv + 1e-6f);
    }
}

// ------------------------------ GEMM2: R = sub @ Wf (split-K over d) -------
// grid: (8 Mtiles, 8 devices). block: 256 (8 waves, 16 rows each, N=32).
// A and B tiles staged by async DMA (both already bf16 in global).
__global__ __launch_bounds__(256) void k_gemm2(
    const unsigned short* __restrict__ sub,
    const unsigned short* __restrict__ Wf, float* __restrict__ R)
{
    __shared__ unsigned short As[128 * 32];  // 8 KB
    __shared__ unsigned short Bs[32 * 32];   // [j][k], 2 KB
    const int d  = blockIdx.y;
    const int m0 = blockIdx.x * 128;
    const int tid = threadIdx.x;
    const int wave = tid >> 5, lane = tid & 31;
    const int half = lane >> 4, l16 = lane & 15;

    v8f acc[2] = {};
    for (int ks = 0; ks < NXT; ks += 32) {
        __syncthreads();   // previous tile fully consumed before overwrite
        // As: 128 rows x 64B = 512 x 16B transfers, 2 per thread
#pragma unroll
        for (int t = 0; t < 2; ++t) {
            int idx = t * 256 + tid;
            int r = idx >> 2, g = idx & 3;
            async_b128((unsigned)(uintptr_t)&As[r * 32 + g * 8], sub,
                       (unsigned)(((unsigned)(m0 + r) * KTOT + d * NXT + ks + g * 8) * 2));
        }
        // Bs: 32 rows (j) x 64B = 128 x 16B transfers (threads 0..127)
        if (tid < 128) {
            int j = tid >> 2, g = tid & 3;
            async_b128((unsigned)(uintptr_t)&Bs[j * 32 + g * 8], Wf,
                       (unsigned)(((d * 32 + j) * NXT + ks + g * 8) * 2));
        }
        async_wait0();
        __syncthreads();

        v16u a, bm[2];
        int row = wave * 16 + l16;
#pragma unroll
        for (int v = 0; v < 8; ++v) {
            int kk = (v < 4) ? (8 * half + 2 * v) : (16 + 8 * half + 2 * (v - 4));
            unsigned int t = *(const unsigned int*)&As[row * 32 + kk];
            a[2 * v]     = (unsigned short)t;
            a[2 * v + 1] = (unsigned short)(t >> 16);
        }
#pragma unroll
        for (int ni = 0; ni < 2; ++ni) {
            int j = ni * 16 + l16;
#pragma unroll
            for (int v = 0; v < 8; ++v) {
                int kk = (v < 4) ? (8 * half + 2 * v) : (16 + 8 * half + 2 * (v - 4));
                unsigned int t = *(const unsigned int*)&Bs[j * 32 + kk];
                bm[ni][2 * v]     = (unsigned short)t;
                bm[ni][2 * v + 1] = (unsigned short)(t >> 16);
            }
        }
#pragma unroll
        for (int ni = 0; ni < 2; ++ni)
            acc[ni] = __builtin_amdgcn_wmma_f32_16x16x32_bf16(
                false, __builtin_bit_cast(v16bf, a),
                false, __builtin_bit_cast(v16bf, bm[ni]),
                (short)0, acc[ni], false, false);
    }
#pragma unroll
    for (int ni = 0; ni < 2; ++ni) {
        int j = ni * 16 + l16;
#pragma unroll
        for (int r = 0; r < 8; ++r) {
            int m = m0 + wave * 16 + 8 * half + r;
            atomicAdd(&R[m * NSYM + j], acc[ni][r]);
        }
    }
}

// ------------------------------ final MLP + log_softmax --------------------
__global__ __launch_bounds__(32) void k_final(
    const float* __restrict__ R, const float* __restrict__ cAB,
    const float* __restrict__ sc,
    const float* __restrict__ W3, const float* __restrict__ b3,
    const float* __restrict__ W4, const float* __restrict__ b4,
    const float* __restrict__ Wo, const float* __restrict__ bo,
    float* __restrict__ out)
{
    __shared__ float sW3[98 * 32], sb3[98], sW4[49 * 98], sb4[49];
    __shared__ float sWo[10 * 49], sbo[10];
    __shared__ float h1[32][32];
    __shared__ float h2[32][98];
    __shared__ float h3[32][49];
    const int tid = threadIdx.x;

    for (int i = tid; i < 98 * 32; i += 32) sW3[i] = W3[i];
    for (int i = tid; i < 98;      i += 32) sb3[i] = b3[i];
    for (int i = tid; i < 49 * 98; i += 32) sW4[i] = W4[i];
    for (int i = tid; i < 49;      i += 32) sb4[i] = b4[i];
    for (int i = tid; i < 10 * 49; i += 32) sWo[i] = Wo[i];
    for (int i = tid; i < 10;      i += 32) sbo[i] = bo[i];
    __syncthreads();

    const int b = blockIdx.x * 32 + tid;
    if (b >= BATCH) return;
    const float gm = sc[0], inv_s = sc[1];

    for (int j = 0; j < 32; ++j) {
        float pre = inv_s * R[b * 32 + j] + cAB[j] - gm * inv_s * cAB[32 + j];
        h1[tid][j] = pre > 0.0f ? pre : 0.0f;
    }
    for (int i = 0; i < 98; ++i) {
        float s = sb3[i];
        for (int j = 0; j < 32; ++j) s += h1[tid][j] * sW3[i * 32 + j];
        h2[tid][i] = s > 0.0f ? s : 0.0f;
    }
    for (int i = 0; i < 49; ++i) {
        float s = sb4[i];
        for (int j = 0; j < 98; ++j) s += h2[tid][j] * sW4[i * 98 + j];
        h3[tid][i] = s > 0.0f ? s : 0.0f;
    }
    float lg[10];
    float mx = -1e30f;
#pragma unroll
    for (int o = 0; o < 10; ++o) {
        float s = sbo[o];
        for (int j = 0; j < 49; ++j) s += h3[tid][j] * sWo[o * 49 + j];
        lg[o] = s;
        mx = s > mx ? s : mx;
    }
    float se = 0.0f;
#pragma unroll
    for (int o = 0; o < 10; ++o) se += __expf(lg[o] - mx);
    float lse = __logf(se);
#pragma unroll
    for (int o = 0; o < 10; ++o) out[b * 10 + o] = lg[o] - mx - lse;
}

// ---------------------------------------------------------------------------
extern "C" void kernel_launch(void* const* d_in, const int* in_sizes, int n_in,
                              void* d_out, int out_size, void* d_ws, size_t ws_size,
                              hipStream_t stream) {
    (void)in_sizes; (void)n_in; (void)out_size; (void)ws_size;
    const float* x     = (const float*)d_in[0];
    const float* W1    = (const float*)d_in[1];
    const float* b1    = (const float*)d_in[2];
    const float* Wc    = (const float*)d_in[3];
    const float* bc    = (const float*)d_in[4];
    const float* cut   = (const float*)d_in[5];
    const float* b_mat = (const float*)d_in[6];
    const float* h_mat = (const float*)d_in[7];
    const float* a_mat = (const float*)d_in[8];
    const float* noise = (const float*)d_in[9];
    const float* W3    = (const float*)d_in[10];
    const float* b3    = (const float*)d_in[11];
    const float* W4    = (const float*)d_in[12];
    const float* b4    = (const float*)d_in[13];
    const float* Wo    = (const float*)d_in[14];
    const float* bo    = (const float*)d_in[15];
    const float* ind   = (const float*)d_in[16];

    // workspace layout (all 16B aligned)
    char* ws = (char*)d_ws;
    unsigned short* sub  = (unsigned short*)ws;                       // 32 MiB
    size_t off = (size_t)MPAD * KTOT * 2;
    unsigned short* Wf   = (unsigned short*)(ws + off);  off += (size_t)KTOT * NSYM * 2;
    unsigned short* xbf  = (unsigned short*)(ws + off);  off += (size_t)XE * 2;
    unsigned short* w1bf = (unsigned short*)(ws + off);  off += (size_t)WE * 2;
    float* R   = (float*)(ws + off);
    float* S12 = R + MPAD * NSYM;   // 16 floats
    float* cAB = S12 + 16;          // 64 floats
    float* sc  = cAB + 64;          // 2 floats

    const int nz = MPAD * NSYM + 16 + 64 + 2;
    hipLaunchKernelGGL(k_zero, dim3((nz + 255) / 256), dim3(256), 0, stream, R, nz);
    hipLaunchKernelGGL(k_conv, dim3((XE + WE + 255) / 256), dim3(256), 0, stream,
                       x, W1, xbf, w1bf);
    hipLaunchKernelGGL(k_prep, dim3(256), dim3(64), 0, stream,
                       Wc, bc, cut, b_mat, h_mat, a_mat, noise, ind, Wf, cAB);
    hipLaunchKernelGGL(k_gemm1, dim3(8, 32, 8), dim3(256), 0, stream,
                       xbf, w1bf, b1, sub, S12);
    hipLaunchKernelGGL(k_stats, dim3(1), dim3(32), 0, stream, S12, sc);
    hipLaunchKernelGGL(k_gemm2, dim3(8, 8), dim3(256), 0, stream, sub, Wf, R);
    hipLaunchKernelGGL(k_final, dim3(32), dim3(32), 0, stream,
                       R, cAB, sc, W3, b3, W4, b4, Wo, bo, (float*)d_out);
}
